// DeepST_model_7052336300359
// MI455X (gfx1250) — compile-verified, matrix operand order checked by reading, not run on hardware
//
#include <hip/hip_runtime.h>

// ---------------------------------------------------------------------------
// DeepST forward on MI455X (gfx1250, wave32, WMMA)
// Big GEMMs (3000->32 and 32->3000) use v_wmma_f32_16x16x32_f16 (HBM-bound,
// f16 A/B + f32 accumulate). Everything else is small VALU/atomic kernels.
// ---------------------------------------------------------------------------

typedef __attribute__((ext_vector_type(16))) _Float16 v16h;
typedef __attribute__((ext_vector_type(8)))  float    v8f;

union V16u {
    v16h     v;
    _Float16 h[16];
    unsigned u[8];
};

#define NNODE   40000
#define DIN     3000
#define EEDGE   640000
#define NE      680000     // E + N self loops
#define KPAD0   3008       // K of enc0 padded to 94*32
#define NPAD    3008       // N of out GEMM padded to 188*16

__device__ __forceinline__ float eluf(float x)   { return x > 0.f ? x : expm1f(x); }
__device__ __forceinline__ float lreluf(float x) { return x > 0.f ? x : 0.2f * x; }

// order-preserving float <-> uint mapping for atomicMax on floats
__device__ __forceinline__ unsigned fmap(float f) {
    int i = __float_as_int(f);
    return (unsigned)(i ^ ((i >> 31) | 0x80000000));
}
__device__ __forceinline__ float funmap(unsigned u) {
    int i = (u & 0x80000000u) ? (int)(u ^ 0x80000000u) : ~(int)u;
    return __int_as_float(i);
}

// K position of packed f16 pair for WMMA A/B register layout:
// vgpr v in [0,8), lane-group g in {0,1}  ->  pair (k, k+1)
__device__ __forceinline__ int kpos(int v, int g) {
    return ((v & 4) << 2) + ((v & 3) << 1) + 8 * g;
}

// pack four float4s (two contiguous 32B segments) into the 16-half A/B operand
__device__ __forceinline__ void pack16(V16u& A, float4 q0, float4 q1, float4 q2, float4 q3) {
    A.h[0]  = (_Float16)q0.x; A.h[1]  = (_Float16)q0.y;
    A.h[2]  = (_Float16)q0.z; A.h[3]  = (_Float16)q0.w;
    A.h[4]  = (_Float16)q1.x; A.h[5]  = (_Float16)q1.y;
    A.h[6]  = (_Float16)q1.z; A.h[7]  = (_Float16)q1.w;
    A.h[8]  = (_Float16)q2.x; A.h[9]  = (_Float16)q2.y;
    A.h[10] = (_Float16)q2.z; A.h[11] = (_Float16)q2.w;
    A.h[12] = (_Float16)q3.x; A.h[13] = (_Float16)q3.y;
    A.h[14] = (_Float16)q3.z; A.h[15] = (_Float16)q3.w;
}

// ---------------------------------------------------------------------------
// zero fill (grid-stride)
// ---------------------------------------------------------------------------
__global__ void zero_kernel(float* p, long n) {
    long t = (long)blockIdx.x * blockDim.x + threadIdx.x;
    long s = (long)gridDim.x * blockDim.x;
    for (; t < n; t += s) p[t] = 0.f;
}

// ---------------------------------------------------------------------------
// weight swizzle: e0_w [3000,32] -> WMMA B layout, f16 pairs, K padded to 3008
// one thread per (chunk, ntile, lane) -> 8 dwords
// ---------------------------------------------------------------------------
__global__ void swz_w0(const float* __restrict__ w, unsigned* __restrict__ o) {
    int t = blockIdx.x * blockDim.x + threadIdx.x;
    if (t >= (KPAD0 / 32) * 2 * 32) return;
    int lane = t & 31;
    int nt   = (t >> 5) & 1;
    int c    = t >> 6;
    int g = lane >> 4;
    int n = nt * 16 + (lane & 15);
    unsigned* dst = o + (size_t)t * 8;
    for (int v = 0; v < 8; ++v) {
        int k = c * 32 + kpos(v, g);
        float f0 = (k     < DIN) ? w[(size_t)k       * 32 + n] : 0.f;
        float f1 = (k + 1 < DIN) ? w[(size_t)(k + 1) * 32 + n] : 0.f;
        union { _Float16 h[2]; unsigned u; } pk;
        pk.h[0] = (_Float16)f0; pk.h[1] = (_Float16)f1;
        dst[v] = pk.u;
    }
}

// out_w [32,3000] -> WMMA B layout, N padded to 3008, K=32 (single chunk)
__global__ void swz_wout(const float* __restrict__ w, unsigned* __restrict__ o) {
    int t = blockIdx.x * blockDim.x + threadIdx.x;
    if (t >= (NPAD / 16) * 32) return;
    int lane = t & 31;
    int nt   = t >> 5;
    int g = lane >> 4;
    int n = nt * 16 + (lane & 15);
    unsigned* dst = o + (size_t)t * 8;
    for (int v = 0; v < 8; ++v) {
        int k = kpos(v, g);
        float f0 = (n < DIN) ? w[(size_t)k       * DIN + n] : 0.f;
        float f1 = (n < DIN) ? w[(size_t)(k + 1) * DIN + n] : 0.f;
        union { _Float16 h[2]; unsigned u; } pk;
        pk.h[0] = (_Float16)f0; pk.h[1] = (_Float16)f1;
        dst[v] = pk.u;
    }
}

// ---------------------------------------------------------------------------
// encoder GEMM0: h = ELU(BN(x @ e0_w + b))   x:[40000,3000]  -> h:[40000,32]
// 4 waves/block, one 16-row tile per wave, two 16-wide accumulators.
// Main K loop (chunks 0..92) has no boundary predicates; tail chunk 93 zeroes
// only the ag==1 lanes' second segment (k >= 3000).
// ---------------------------------------------------------------------------
__global__ __launch_bounds__(128)
void gemm_enc0(const float* __restrict__ x, const unsigned* __restrict__ w0s,
               const float* __restrict__ b, const float* __restrict__ g,
               const float* __restrict__ beta, const float* __restrict__ m,
               const float* __restrict__ vv_, float* __restrict__ hout) {
    int wid  = threadIdx.x >> 5;
    int lane = threadIdx.x & 31;
    int row0 = (blockIdx.x * 4 + wid) * 16;
    int am = lane & 15, ag = lane >> 4;
    // per-lane A base: two contiguous 32B segments at +0 and +16 floats
    const float* xseg = x + (size_t)(row0 + am) * DIN + 8 * ag;

    v8f c0 = {}; v8f c1 = {};
    const unsigned* pb = w0s + (size_t)lane * 8;

    for (int c = 0; c < 93; ++c) {
        const float* p = xseg + c * 32;
        float4 q0 = ((const float4*)p)[0];
        float4 q1 = ((const float4*)p)[1];
        float4 q2 = ((const float4*)(p + 16))[0];
        float4 q3 = ((const float4*)(p + 16))[1];
        V16u A; pack16(A, q0, q1, q2, q3);

        V16u B0, B1;
        const unsigned* p0 = pb + (size_t)(c * 2 + 0) * 256;
        const unsigned* p1 = pb + (size_t)(c * 2 + 1) * 256;
        #pragma unroll
        for (int v = 0; v < 8; ++v) { B0.u[v] = p0[v]; B1.u[v] = p1[v]; }
        c0 = __builtin_amdgcn_wmma_f32_16x16x32_f16(false, A.v, false, B0.v, (short)0, c0, false, false);
        c1 = __builtin_amdgcn_wmma_f32_16x16x32_f16(false, A.v, false, B1.v, (short)0, c1, false, false);
    }
    { // tail chunk c = 93: k in [2976, 3008); only ag==1 second segment is OOB
        const float* p = xseg + 93 * 32;
        float4 q0 = ((const float4*)p)[0];
        float4 q1 = ((const float4*)p)[1];
        float4 q2 = make_float4(0.f, 0.f, 0.f, 0.f);
        float4 q3 = q2;
        if (ag == 0) {
            q2 = ((const float4*)(p + 16))[0];
            q3 = ((const float4*)(p + 16))[1];
        }
        V16u A; pack16(A, q0, q1, q2, q3);

        V16u B0, B1;
        const unsigned* p0 = pb + (size_t)(93 * 2 + 0) * 256;
        const unsigned* p1 = pb + (size_t)(93 * 2 + 1) * 256;
        #pragma unroll
        for (int v = 0; v < 8; ++v) { B0.u[v] = p0[v]; B1.u[v] = p1[v]; }
        c0 = __builtin_amdgcn_wmma_f32_16x16x32_f16(false, A.v, false, B0.v, (short)0, c0, false, false);
        c1 = __builtin_amdgcn_wmma_f32_16x16x32_f16(false, A.v, false, B1.v, (short)0, c1, false, false);
    }

    // fused BN(eps=1e-3) + ELU epilogue.  D layout: vgpr r -> M=r (+8 for hi lanes), N=lane%16
    int n0 = am, n1 = 16 + am;
    float s0 = g[n0] * rsqrtf(vv_[n0] + 1e-3f);
    float o0 = (b[n0] - m[n0]) * s0 + beta[n0];
    float s1 = g[n1] * rsqrtf(vv_[n1] + 1e-3f);
    float o1 = (b[n1] - m[n1]) * s1 + beta[n1];
    #pragma unroll
    for (int r = 0; r < 8; ++r) {
        int mm = row0 + r + 8 * ag;
        hout[(size_t)mm * 32 + n0] = eluf(c0[r] * s0 + o0);
        hout[(size_t)mm * 32 + n1] = eluf(c1[r] * s1 + o1);
    }
}

// ---------------------------------------------------------------------------
// encoder L1: feat_x = ELU(BN(h @ e1_w + b))   32 -> 20
// ---------------------------------------------------------------------------
__global__ __launch_bounds__(256)
void enc1_kernel(const float* __restrict__ h, const float* __restrict__ w,
                 const float* __restrict__ b, const float* __restrict__ g,
                 const float* __restrict__ beta, const float* __restrict__ m,
                 const float* __restrict__ vv_, float* __restrict__ fx) {
    int i = blockIdx.x * blockDim.x + threadIdx.x;
    if (i >= NNODE) return;
    float hv[32];
    #pragma unroll
    for (int k = 0; k < 32; ++k) hv[k] = h[(size_t)i * 32 + k];
    #pragma unroll 4
    for (int j = 0; j < 20; ++j) {
        float acc = b[j];
        #pragma unroll
        for (int k = 0; k < 32; ++k) acc += hv[k] * w[k * 20 + j];
        float s = g[j] * rsqrtf(vv_[j] + 1e-3f);
        fx[(size_t)i * 20 + j] = eluf((acc - m[j]) * s + beta[j]);
    }
}

// ---------------------------------------------------------------------------
// GAT linear transform + attention scalars (templated on in/out width)
// ---------------------------------------------------------------------------
template <int NIN, int NOUT>
__global__ __launch_bounds__(256)
void gat_hw(const float* __restrict__ in, const float* __restrict__ w,
            const float* __restrict__ as_, const float* __restrict__ ad_,
            float* __restrict__ hw, float* __restrict__ ssrc, float* __restrict__ sdst) {
    __shared__ float sw[NIN * NOUT];
    __shared__ float sas[NOUT], sad[NOUT];
    for (int t = threadIdx.x; t < NIN * NOUT; t += 256) sw[t] = w[t];
    if (threadIdx.x < NOUT) { sas[threadIdx.x] = as_[threadIdx.x]; sad[threadIdx.x] = ad_[threadIdx.x]; }
    __syncthreads();
    int i = blockIdx.x * blockDim.x + threadIdx.x;
    if (i >= NNODE) return;
    float xv[NIN];
    #pragma unroll
    for (int k = 0; k < NIN; ++k) xv[k] = in[(size_t)i * NIN + k];
    float s1 = 0.f, s2 = 0.f;
    for (int j = 0; j < NOUT; ++j) {
        float acc = 0.f;
        #pragma unroll
        for (int k = 0; k < NIN; ++k) acc += xv[k] * sw[k * NOUT + j];
        hw[(size_t)i * NOUT + j] = acc;
        s1 += acc * sas[j];
        s2 += acc * sad[j];
    }
    ssrc[i] = s1; sdst[i] = s2;
}

// ---------------------------------------------------------------------------
// edge passes for segment softmax
// ---------------------------------------------------------------------------
__device__ __forceinline__ void edge_sd(const int* ei, int e, int& s, int& d) {
    if (e < EEDGE) { s = ei[e]; d = ei[EEDGE + e]; }
    else           { s = e - EEDGE; d = s; }
}

__global__ void edge_max_kernel(const int* __restrict__ ei, const float* __restrict__ ssrc,
                                const float* __restrict__ sdst, float* __restrict__ alE,
                                unsigned* __restrict__ mx) {
    int e = blockIdx.x * blockDim.x + threadIdx.x;
    if (e >= NE) return;
    int s, d; edge_sd(ei, e, s, d);
    float al = lreluf(ssrc[s] + sdst[d]);
    alE[e] = al;
    atomicMax(&mx[d], fmap(al));
}

__global__ void edge_exp_kernel(const int* __restrict__ ei, float* __restrict__ alE,
                                const unsigned* __restrict__ mx, float* __restrict__ zs) {
    int e = blockIdx.x * blockDim.x + threadIdx.x;
    if (e >= NE) return;
    int s, d; edge_sd(ei, e, s, d);
    float ex = expf(alE[e] - funmap(mx[d]));
    alE[e] = ex;
    atomicAdd(&zs[d], ex);
}

template <int F>
__global__ void edge_agg_kernel(const int* __restrict__ ei, const float* __restrict__ alE,
                                const float* __restrict__ zs, const float* __restrict__ hw,
                                float* __restrict__ agg) {
    const int GP = F / 4;
    int tid = blockIdx.x * blockDim.x + threadIdx.x;
    int e = tid / GP;
    if (e >= NE) return;
    int f0 = (tid % GP) * 4;
    int s, d; edge_sd(ei, e, s, d);
    float w = alE[e] / (zs[d] + 1e-16f);
    const float* hs = hw + (size_t)s * F + f0;
    float* ad = agg + (size_t)d * F + f0;
    #pragma unroll
    for (int j = 0; j < 4; ++j) atomicAdd(&ad[j], w * hs[j]);
}

// conv1 post: c = relu(BN(agg + bias, eps=1e-5)) in place
__global__ void gat_post_kernel(float* __restrict__ c, const float* __restrict__ b,
                                const float* __restrict__ g, const float* __restrict__ beta,
                                const float* __restrict__ m, const float* __restrict__ vv_) {
    int t = blockIdx.x * blockDim.x + threadIdx.x;
    if (t >= NNODE * 64) return;
    int f = t & 63;
    float s = g[f] * rsqrtf(vv_[f] + 1e-5f);
    float val = (c[t] + b[f] - m[f]) * s + beta[f];
    c[t] = val > 0.f ? val : 0.f;
}

__global__ void add_bias8(float* __restrict__ o, const float* __restrict__ b) {
    int t = blockIdx.x * blockDim.x + threadIdx.x;
    if (t >= NNODE * 8) return;
    o[t] += b[t & 7];
}

// gnn_z = eps * exp(logvar) + mu ; z = [feat_x | gnn_z]
__global__ void reparam_kernel(const float* __restrict__ eps, const float* __restrict__ mu,
                               const float* __restrict__ lv, const float* __restrict__ fx,
                               float* __restrict__ gz, float* __restrict__ z) {
    int t = blockIdx.x * blockDim.x + threadIdx.x;
    if (t >= NNODE * 28) return;
    int i = t / 28, j = t - i * 28;
    if (j < 20) {
        z[t] = fx[(size_t)i * 20 + j];
    } else {
        int f = j - 20;
        float v = eps[(size_t)i * 8 + f] * expf(lv[(size_t)i * 8 + f]) + mu[(size_t)i * 8 + f];
        gz[(size_t)i * 8 + f] = v;
        z[t] = v;
    }
}

// decoder L0: d = ELU(BN(z @ d0_w + b, eps=1e-3))   28 -> 32
__global__ __launch_bounds__(256)
void dec0_kernel(const float* __restrict__ z, const float* __restrict__ w,
                 const float* __restrict__ b, const float* __restrict__ g,
                 const float* __restrict__ beta, const float* __restrict__ m,
                 const float* __restrict__ vv_, float* __restrict__ dh) {
    int i = blockIdx.x * blockDim.x + threadIdx.x;
    if (i >= NNODE) return;
    float zv[28];
    #pragma unroll
    for (int k = 0; k < 28; ++k) zv[k] = z[(size_t)i * 28 + k];
    #pragma unroll 4
    for (int j = 0; j < 32; ++j) {
        float acc = b[j];
        #pragma unroll
        for (int k = 0; k < 28; ++k) acc += zv[k] * w[k * 32 + j];
        float s = g[j] * rsqrtf(vv_[j] + 1e-3f);
        dh[(size_t)i * 32 + j] = eluf((acc - m[j]) * s + beta[j]);
    }
}

// ---------------------------------------------------------------------------
// decoder GEMM: de_feat = dh @ out_w + out_b    [40000,32] x [32,3000]
// K=32 = exactly one WMMA chunk; A loaded once per wave, sweep N tiles
// ---------------------------------------------------------------------------
__global__ __launch_bounds__(128)
void gemm_out(const float* __restrict__ dh, const unsigned* __restrict__ wos,
              const float* __restrict__ ob, float* __restrict__ out) {
    int wid  = threadIdx.x >> 5;
    int lane = threadIdx.x & 31;
    int row0 = (blockIdx.x * 4 + wid) * 16;
    int am = lane & 15, ag = lane >> 4;

    const float* p = dh + (size_t)(row0 + am) * 32 + 8 * ag;
    float4 q0 = ((const float4*)p)[0];
    float4 q1 = ((const float4*)p)[1];
    float4 q2 = ((const float4*)(p + 16))[0];
    float4 q3 = ((const float4*)(p + 16))[1];
    V16u A; pack16(A, q0, q1, q2, q3);

    int t0 = blockIdx.y * 47;       // 4 * 47 = 188 tiles cover NPAD=3008
    for (int t = t0; t < t0 + 47; ++t) {
        V16u B;
        const unsigned* pw = wos + ((size_t)t * 32 + lane) * 8;
        #pragma unroll
        for (int v = 0; v < 8; ++v) B.u[v] = pw[v];
        v8f c = {};
        c = __builtin_amdgcn_wmma_f32_16x16x32_f16(false, A.v, false, B.v, (short)0, c, false, false);
        int n = t * 16 + am;
        if (n < DIN) {
            float bb = ob[n];
            #pragma unroll
            for (int r = 0; r < 8; ++r)
                out[(size_t)(row0 + r + 8 * ag) * DIN + n] = c[r] + bb;
        }
    }
}

// ---------------------------------------------------------------------------
// DEC student-t soft assignment
// ---------------------------------------------------------------------------
__global__ __launch_bounds__(256)
void q_kernel(const float* __restrict__ z, const float* __restrict__ cl, float* __restrict__ q) {
    __shared__ float sc[15 * 28];
    for (int t = threadIdx.x; t < 15 * 28; t += 256) sc[t] = cl[t];
    __syncthreads();
    int i = blockIdx.x * blockDim.x + threadIdx.x;
    if (i >= NNODE) return;
    float zv[28];
    #pragma unroll
    for (int j = 0; j < 28; ++j) zv[j] = z[(size_t)i * 28 + j];
    float qs[15]; float sum = 0.f;
    for (int k = 0; k < 15; ++k) {
        float sq = 0.f;
        #pragma unroll
        for (int j = 0; j < 28; ++j) { float df = zv[j] - sc[k * 28 + j]; sq += df * df; }
        float qq = powf(1.f + sq / 0.9f + 1e-8f, -0.95f);
        qs[k] = qq; sum += qq;
    }
    float inv = 1.f / sum;
    for (int k = 0; k < 15; ++k) q[(size_t)i * 15 + k] = qs[k] * inv;
}

// ---------------------------------------------------------------------------
// host-side orchestration
// ---------------------------------------------------------------------------
extern "C" void kernel_launch(void* const* d_in, const int* in_sizes, int n_in,
                              void* d_out, int out_size, void* d_ws, size_t ws_size,
                              hipStream_t stream) {
    const float* x       = (const float*)d_in[0];
    const int*   ei      = (const int*)  d_in[1];
    const float* e0_w    = (const float*)d_in[2];
    const float* e0_b    = (const float*)d_in[3];
    const float* e0_g    = (const float*)d_in[4];
    const float* e0_beta = (const float*)d_in[5];
    const float* e0_m    = (const float*)d_in[6];
    const float* e0_v    = (const float*)d_in[7];
    const float* e1_w    = (const float*)d_in[8];
    const float* e1_b    = (const float*)d_in[9];
    const float* e1_g    = (const float*)d_in[10];
    const float* e1_beta = (const float*)d_in[11];
    const float* e1_m    = (const float*)d_in[12];
    const float* e1_v    = (const float*)d_in[13];
    const float* g1_w    = (const float*)d_in[14];
    const float* g1_as   = (const float*)d_in[15];
    const float* g1_ad   = (const float*)d_in[16];
    const float* g1_b    = (const float*)d_in[17];
    const float* bnc_g   = (const float*)d_in[18];
    const float* bnc_beta= (const float*)d_in[19];
    const float* bnc_m   = (const float*)d_in[20];
    const float* bnc_v   = (const float*)d_in[21];
    const float* gm_w    = (const float*)d_in[22];
    const float* gm_as   = (const float*)d_in[23];
    const float* gm_ad   = (const float*)d_in[24];
    const float* gm_b    = (const float*)d_in[25];
    const float* gv_w    = (const float*)d_in[26];
    const float* gv_as   = (const float*)d_in[27];
    const float* gv_ad   = (const float*)d_in[28];
    const float* gv_b    = (const float*)d_in[29];
    const float* d0_w    = (const float*)d_in[30];
    const float* d0_b    = (const float*)d_in[31];
    const float* d0_g    = (const float*)d_in[32];
    const float* d0_beta = (const float*)d_in[33];
    const float* d0_m    = (const float*)d_in[34];
    const float* d0_v    = (const float*)d_in[35];
    const float* out_w   = (const float*)d_in[36];
    const float* out_b   = (const float*)d_in[37];
    const float* cluster = (const float*)d_in[38];
    const float* eps     = (const float*)d_in[39];

    float* out  = (float*)d_out;
    float* o_z  = out;                         // 40000*28
    float* o_mu = out + 1120000;               // 40000*8
    float* o_lv = out + 1440000;               // 40000*8
    float* o_de = out + 1760000;               // 40000*3000
    float* o_q  = out + 121760000;             // 40000*15
    float* o_fx = out + 122360000;             // 40000*20
    float* o_gz = out + 123160000;             // 40000*8

    // workspace arena (floats)
    float*    ws  = (float*)d_ws;
    unsigned* w0s = (unsigned*)ws;             // 48128 u32  (enc0 weights swizzled)
    unsigned* wos = (unsigned*)(ws + 48128);   // 48128 u32  (out weights swizzled)
    float* h   = ws + 96256;                   // 40000*32
    float* hw  = h   + 1280000;                // 40000*64 (reused for 40000*8)
    float* agg = hw  + 2560000;                // 40000*64 (becomes c)
    float* ss  = agg + 2560000;                // 40000
    float* sd  = ss  + 40000;                  // 40000
    float* alE = sd  + 40000;                  // 680000
    unsigned* mx = (unsigned*)(alE + 680000);  // 40000
    float* zs  = (float*)mx + 40000;           // 40000
    float* dh  = zs  + 40000;                  // 40000*32

    const int B = 256;
    dim3 gN((NNODE + B - 1) / B);
    dim3 gE((NE + B - 1) / B);

    // ---- weight prep (deterministic, rerun every call) ----
    swz_w0  <<<(6016 + B - 1) / B, B, 0, stream>>>(e0_w, w0s);
    swz_wout<<<(6016 + B - 1) / B, B, 0, stream>>>(out_w, wos);

    // ---- encoder ----
    gemm_enc0<<<625, 128, 0, stream>>>(x, w0s, e0_b, e0_g, e0_beta, e0_m, e0_v, h);
    enc1_kernel<<<gN, B, 0, stream>>>(h, e1_w, e1_b, e1_g, e1_beta, e1_m, e1_v, o_fx);

    // ---- GAT conv1 (20 -> 64) ----
    gat_hw<20, 64><<<gN, B, 0, stream>>>(o_fx, g1_w, g1_as, g1_ad, hw, ss, sd);
    zero_kernel<<<2048, B, 0, stream>>>((float*)mx, 40000);
    zero_kernel<<<2048, B, 0, stream>>>(zs, 40000);
    zero_kernel<<<2048, B, 0, stream>>>(agg, (long)NNODE * 64);
    edge_max_kernel<<<gE, B, 0, stream>>>(ei, ss, sd, alE, mx);
    edge_exp_kernel<<<gE, B, 0, stream>>>(ei, alE, mx, zs);
    edge_agg_kernel<64><<<((long)NE * 16 + B - 1) / B, B, 0, stream>>>(ei, alE, zs, hw, agg);
    gat_post_kernel<<<((long)NNODE * 64 + B - 1) / B, B, 0, stream>>>(agg, g1_b, bnc_g, bnc_beta, bnc_m, bnc_v);

    // ---- GAT mu (64 -> 8), aggregate directly into d_out mu region ----
    gat_hw<64, 8><<<gN, B, 0, stream>>>(agg, gm_w, gm_as, gm_ad, hw, ss, sd);
    zero_kernel<<<2048, B, 0, stream>>>((float*)mx, 40000);
    zero_kernel<<<2048, B, 0, stream>>>(zs, 40000);
    zero_kernel<<<2048, B, 0, stream>>>(o_mu, (long)NNODE * 8);
    edge_max_kernel<<<gE, B, 0, stream>>>(ei, ss, sd, alE, mx);
    edge_exp_kernel<<<gE, B, 0, stream>>>(ei, alE, mx, zs);
    edge_agg_kernel<8><<<((long)NE * 2 + B - 1) / B, B, 0, stream>>>(ei, alE, zs, hw, o_mu);
    add_bias8<<<((long)NNODE * 8 + B - 1) / B, B, 0, stream>>>(o_mu, gm_b);

    // ---- GAT logvar (64 -> 8) ----
    gat_hw<64, 8><<<gN, B, 0, stream>>>(agg, gv_w, gv_as, gv_ad, hw, ss, sd);
    zero_kernel<<<2048, B, 0, stream>>>((float*)mx, 40000);
    zero_kernel<<<2048, B, 0, stream>>>(zs, 40000);
    zero_kernel<<<2048, B, 0, stream>>>(o_lv, (long)NNODE * 8);
    edge_max_kernel<<<gE, B, 0, stream>>>(ei, ss, sd, alE, mx);
    edge_exp_kernel<<<gE, B, 0, stream>>>(ei, alE, mx, zs);
    edge_agg_kernel<8><<<((long)NE * 2 + B - 1) / B, B, 0, stream>>>(ei, alE, zs, hw, o_lv);
    add_bias8<<<((long)NNODE * 8 + B - 1) / B, B, 0, stream>>>(o_lv, gv_b);

    // ---- reparameterize + concat z ----
    reparam_kernel<<<((long)NNODE * 28 + B - 1) / B, B, 0, stream>>>(eps, o_mu, o_lv, o_fx, o_gz, o_z);

    // ---- decoder ----
    dec0_kernel<<<gN, B, 0, stream>>>(o_z, d0_w, d0_b, d0_g, d0_beta, d0_m, d0_v, dh);
    gemm_out<<<dim3(625, 4), 128, 0, stream>>>(dh, wos, out_b, o_de);

    // ---- cluster soft assignment ----
    q_kernel<<<gN, B, 0, stream>>>(o_z, cluster, o_q);
}
